// GaussianImage_Covariance_83081847374187
// MI455X (gfx1250) — matrix operand order repaired and checked by reading, not compiled, not run on hardware
//
#include <hip/hip_runtime.h>
#include <math.h>

#define N_PTS      2048
#define IMG_H      512
#define IMG_W      512
#define TILE_G     128          // gaussians staged in LDS per block iteration
#define ALPHA_THR  (1.0f / 255.0f)

typedef __attribute__((ext_vector_type(16))) _Float16 v16h;
typedef __attribute__((ext_vector_type(8)))  float    v8f;

// Raw hardware transcendentals: v_exp_f32 computes 2^x, v_cos_f32 computes
// cos(2*pi*x). We pre-scale parameters so no per-pair range conversion is needed.
#if __has_builtin(__builtin_amdgcn_exp2f)
#define EXP2F(x) __builtin_amdgcn_exp2f(x)
#else
#define EXP2F(x) exp2f(x)
#endif
#if __has_builtin(__builtin_amdgcn_cosf)
#define COS_REV(x) __builtin_amdgcn_cosf(x)                       // cos(2*pi*x)
#else
#define COS_REV(x) __cosf((x) * 6.28318530717958647692f)
#endif

#define LOG2E_F   1.44269504088896340736f
#define INV2PI_F  0.15915494309189533577f

// -------- async global->LDS staging (CDNA5), with safe fallback -------------
#if __has_builtin(__builtin_amdgcn_global_load_async_to_lds_b128) && \
    __has_builtin(__builtin_amdgcn_s_wait_asynccnt)
#define USE_ASYNC_LDS 1
// Builtin parameter types (from compiler diagnostic): non-const
// 'int __vector(4) *' in global (AS1 / __device__) and LDS (AS3) spaces.
typedef int v4i_ __attribute__((vector_size(16)));
typedef v4i_ __attribute__((address_space(1))) gv4i;
typedef v4i_ __attribute__((address_space(3))) lv4i;
__device__ __forceinline__ gv4i* to_gbl_(const void* p) {
    return (gv4i*)(unsigned long long)p;
}
__device__ __forceinline__ lv4i* to_lds_(const void* p) {
    // generic LDS pointers carry the LDS byte offset in their low 32 bits
    return (lv4i*)(unsigned int)(unsigned long long)p;
}
#endif

__device__ __forceinline__ float sigmoidf_(float x) {
    return 1.0f / (1.0f + __expf(-x));
}

// ---------------------------------------------------------------------------
// Kernel 1: per-gaussian preprocessing.
//   pA[i*3+0] = {x, y, nca, ncb}        nca = -0.5*conic_a*log2(e), ncb = -conic_b*log2(e)
//   pA[i*3+1] = {ncc, op_eff, gx0, gy0} ncc = -0.5*conic_c*log2(e), g* = freq/(2*pi)
//   pA[i*3+2] = {gx1, gy1, w0, w1}
//   pC[i]     = {r, g, b, 0}
// so that  alpha = op * exp2(nca*dx^2 + ncc*dy^2 + ncb*dx*dy)
//          cos(f.d) = v_cos(gx*dx + gy*dy)
// ---------------------------------------------------------------------------
__global__ __launch_bounds__(256) void gabor_prep_kernel(
    const float* __restrict__ xyz, const float* __restrict__ cov2d,
    const float* __restrict__ fdc, const float* __restrict__ opac,
    const float* __restrict__ gf,  const float* __restrict__ gw,
    float4* __restrict__ pA, float4* __restrict__ pC,
    int* __restrict__ radii, unsigned char* __restrict__ vis)
{
    int i = blockIdx.x * blockDim.x + threadIdx.x;
    if (i >= N_PTS) return;

    float sxx = cov2d[3 * i + 0] + 0.5f;
    float sxy = cov2d[3 * i + 1];
    float syy = cov2d[3 * i + 2] + 0.5f;
    float det = sxx * syy - sxy * sxy;
    bool  valid = det > 1e-8f;
    float det_s = valid ? det : 1.0f;
    float inv   = 1.0f / det_s;
    float ca = syy * inv, cb = -sxy * inv, cc = sxx * inv;

    float mid = 0.5f * (sxx + syy);
    float lam = mid + sqrtf(fmaxf(mid * mid - det, 0.1f));
    int   r   = valid ? (int)ceilf(3.0f * sqrtf(lam)) : 0;
    radii[i] = r;
    vis[i]   = (r > 0) ? 1 : 0;

    float op = valid ? opac[i] : 0.0f;
    if (!valid) { ca = cb = cc = 0.0f; }   // exponent==0 -> alpha==op==0 path

    float nca = -0.5f * ca * LOG2E_F;
    float ncb = -cb * LOG2E_F;
    float ncc = -0.5f * cc * LOG2E_F;

    float c0 = sigmoidf_(fdc[3 * i + 0]);
    float c1 = sigmoidf_(fdc[3 * i + 1]);
    float c2 = sigmoidf_(fdc[3 * i + 2]);

    // gabor_freqs layout: (n*NUM_GABOR, 2), NUM_GABOR == 2; fold 1/(2*pi)
    float gx0 = __expf(gf[(2 * i + 0) * 2 + 0]) * INV2PI_F;
    float gy0 = __expf(gf[(2 * i + 0) * 2 + 1]) * INV2PI_F;
    float gx1 = __expf(gf[(2 * i + 1) * 2 + 0]) * INV2PI_F;
    float gy1 = __expf(gf[(2 * i + 1) * 2 + 1]) * INV2PI_F;
    float w0  = sigmoidf_(gw[2 * i + 0]);
    float w1  = sigmoidf_(gw[2 * i + 1]);

    pA[3 * i + 0] = make_float4(xyz[2 * i + 0], xyz[2 * i + 1], nca, ncb);
    pA[3 * i + 1] = make_float4(ncc, op, gx0, gy0);
    pA[3 * i + 2] = make_float4(gx1, gy1, w0, w1);
    pC[i]         = make_float4(c0, c1, c2, 0.0f);
}

// ---------------------------------------------------------------------------
// Kernel 2: render. One wave32 owns a 16-pixel horizontal strip (M dim).
// einsum('nhw,nc->hwc') runs on v_wmma_f32_16x16x32_f16:
//   A[M][K] = wgt(pixel M, gaussian K)  (f16, computed on the fly)
//   B[K][N] = color channel N of gaussian K (N<3, else 0)
//   D (f32 16x16) accumulates over all 2048 gaussians.
// ---------------------------------------------------------------------------
__global__ __launch_bounds__(256) void gabor_render_wmma_kernel(
    const float4* __restrict__ pA, const float4* __restrict__ pC,
    float* __restrict__ out)
{
    __shared__ float4 sA[TILE_G * 3];
    __shared__ float4 sC[TILE_G];

    const int tid  = threadIdx.x;
    const int lane = tid & 31;
    const int warp = tid >> 5;
    const int wid  = blockIdx.x * 8 + warp;     // 16384 waves total
    const int h     = wid >> 5;                 // 32 strips of 16 px per row
    const int wbase = (wid & 31) << 4;

    const int M  = lane & 15;                   // pixel within strip == A row
    const int hi = lane >> 4;                   // half-wave select
    const int kbA = hi * 8;                     // A: K base per ISA layout
    const int kbB = hi * 16;                    // unused: B indexes K with kbA, like A
    const int N   = M;                          // B/D column (channel)
    const bool hasC = (N < 3);

    const float px = (float)(wbase + M) + 0.5f;
    const float py = (float)h + 0.5f;

    const float* sCf = (const float*)sC;

    v8f acc = {};

    for (int base = 0; base < N_PTS; base += TILE_G) {
        __syncthreads();
        // cooperative stage: 384 float4 of A-params + 128 float4 of colors
        {
            const float4* gA = pA + (size_t)base * 3;
            const int t2 = tid + 256;
#if defined(USE_ASYNC_LDS)
            __builtin_amdgcn_global_load_async_to_lds_b128(
                to_gbl_(gA + tid), to_lds_(&sA[tid]), 0, 0);
            if (t2 < TILE_G * 3) {
                __builtin_amdgcn_global_load_async_to_lds_b128(
                    to_gbl_(gA + t2), to_lds_(&sA[t2]), 0, 0);
            } else {
                __builtin_amdgcn_global_load_async_to_lds_b128(
                    to_gbl_(pC + base + t2 - TILE_G * 3),
                    to_lds_(&sC[t2 - TILE_G * 3]), 0, 0);
            }
            __builtin_amdgcn_s_wait_asynccnt(0);
#else
            sA[tid] = gA[tid];                       // tid in [0,256)
            if (t2 < TILE_G * 3) sA[t2] = gA[t2];
            else                 sC[t2 - TILE_G * 3] = pC[base + t2 - TILE_G * 3];
#endif
        }
        // prefetch next tile into near caches while we compute this one
        if (base + TILE_G < N_PTS) {
            __builtin_prefetch((const char*)(pA + (size_t)(base + TILE_G) * 3) + tid * 24, 0, 1);
            if (tid < 64)
                __builtin_prefetch((const char*)(pC + base + TILE_G) + tid * 32, 0, 1);
        }
        __syncthreads();

        for (int g0 = 0; g0 < TILE_G; g0 += 32) {
            // ---- A tile: 16 wgt values per lane, ISA 16-bit A layout ----
            v16h av;
#pragma unroll
            for (int e = 0; e < 16; ++e) {
                const int k = g0 + kbA + ((e < 8) ? e : (e + 8)); // K = kbA+e / kbA+16+(e-8)
                const float4 p0 = sA[k * 3 + 0];
                const float4 p1 = sA[k * 3 + 1];
                const float4 p2 = sA[k * 3 + 2];
                float dx = px - p0.x;
                float dy = py - p0.y;
                // exponent already folded with -0.5 and log2(e): single v_exp
                float ex = p0.z * dx * dx + p1.x * dy * dy + p0.w * (dx * dy);
                float alpha = p1.y * EXP2F(ex);
                alpha = fminf(alpha, 0.999f);
                alpha = (alpha < ALPHA_THR) ? 0.0f : alpha;
                // frequencies pre-scaled by 1/(2*pi): raw v_cos
                float mod = 1.0f + p2.z * COS_REV(p1.z * dx + p1.w * dy)
                                 + p2.w * COS_REV(p2.x * dx + p2.y * dy);
                av[e] = (_Float16)(alpha * mod);
            }
            // ---- B tile: colors, K striped per half-wave ----
            v16h bv;
#pragma unroll
            for (int t = 0; t < 16; ++t) {
                float c = hasC ? sCf[(g0 + kbA + ((t < 8) ? t : (t + 8))) * 4 + N] : 0.0f;
                bv[t] = (_Float16)c;
            }
            acc = __builtin_amdgcn_wmma_f32_16x16x32_f16(
                false, av, false, bv, (short)0, acc, false, false);
        }
    }

    // ---- write-out: lane holds D[j + 8*hi][N] in acc[j] ----
    if (hasC) {
        float* ch = out + (size_t)N * (IMG_H * IMG_W) + (size_t)h * IMG_W + wbase + hi * 8;
#pragma unroll
        for (int j = 0; j < 8; ++j) {
            float v = acc[j];
            v = fminf(fmaxf(v, 0.0f), 1.0f);
            ch[j] = v;
        }
    }
}

// ---------------------------------------------------------------------------
extern "C" void kernel_launch(void* const* d_in, const int* in_sizes, int n_in,
                              void* d_out, int out_size, void* d_ws, size_t ws_size,
                              hipStream_t stream)
{
    const float* xyz   = (const float*)d_in[0];
    const float* cov2d = (const float*)d_in[1];
    const float* fdc   = (const float*)d_in[2];
    const float* opac  = (const float*)d_in[3];
    const float* gf    = (const float*)d_in[4];
    const float* gw    = (const float*)d_in[5];
    (void)in_sizes; (void)n_in; (void)out_size; (void)ws_size;

    // workspace: packed A-params then colors (128 KB total)
    float4* pA = (float4*)d_ws;
    float4* pC = pA + (size_t)N_PTS * 3;

    // outputs concatenated flat: render(1,3,H,W) f32, radii(int32), visibility(bool)
    float*         render = (float*)d_out;
    char*          outc   = (char*)d_out;
    int*           radii  = (int*)(outc + (size_t)3 * IMG_H * IMG_W * sizeof(float));
    unsigned char* vis    = (unsigned char*)(radii + N_PTS);

    gabor_prep_kernel<<<N_PTS / 256, 256, 0, stream>>>(
        xyz, cov2d, fdc, opac, gf, gw, pA, pC, radii, vis);

    // 512*512 pixels / (16 px per wave * 8 waves per block) = 2048 blocks
    gabor_render_wmma_kernel<<<(IMG_H * IMG_W) / (16 * 8), 256, 0, stream>>>(
        pA, pC, render);
}
